// ACAClassifier_73237782331761
// MI455X (gfx1250) — matrix operand order repaired and checked by reading, compiled, not run on hardware
//
#include <hip/hip_runtime.h>
#include <cstddef>

// ============================================================================
// Gated rule-110 CA classifier, MI455X (gfx1250, wave32, WMMA).
//
// Per step d (8 steps, sequential):
//   mask      = sigmoid(state @ W[d]^T + b[d])        (65536x512x512 GEMM)
//   nb        = floor(4*L + 2*C + R)   (wrap neighbors of current state)
//   bits      = (110 >> nb) & 1
//   new_state = mask*bits + (1-mask)*state
//
// Roofline: 275 GFLOP total; ~2 GB HBM streamed; state+W (136MB) stays in
// the 192MB L2 between steps. bf16 A/B + f32 accumulate via
// v_wmma_f32_16x16x32_bf16 puts compute (~13us/step) at the memory floor;
// the f32 WMMA path (16x16x4, 2048 FLOP/instr) would be ~8x slower.
//
// Tile shape: BM=256 x BN=128 per block (512 thr = 16 wave32s) to cut W
// re-reads from L2 to 256MB/step. Tiles staged global(f32) -> regs ->
// v_cvt_pk_bf16_f32 once -> LDS (pitch 40 bf16, bank-conflict-free),
// double-buffered, one barrier per 32-wide K slab.
// ============================================================================

#define RULE   110
#define CASIZE 512
#define DEPTH  8

#define BM 256          // M tile per block
#define BN 128          // N tile per block
#define BK 32           // K per WMMA / per LDS stage
#define KTILES (CASIZE / BK)
#define LDH 40          // LDS row pitch in bf16 (32 data + 8 pad)

typedef __attribute__((ext_vector_type(16))) __bf16 v16bf;
typedef __attribute__((ext_vector_type(8)))  __bf16 v8bf;
typedef __attribute__((ext_vector_type(8)))  float  v8f;

__device__ __forceinline__ v8bf cvt8(const float4 a, const float4 b) {
    v8bf r;
    r[0] = (__bf16)a.x; r[1] = (__bf16)a.y; r[2] = (__bf16)a.z; r[3] = (__bf16)a.w;
    r[4] = (__bf16)b.x; r[5] = (__bf16)b.y; r[6] = (__bf16)b.z; r[7] = (__bf16)b.w;
    return r;
}

__global__ __launch_bounds__(512)
void aca_step(const float* __restrict__ src,   // current state  [batch,512] f32
              const float* __restrict__ Wd,    // W[d]           [512,512]  f32 (row j = output col)
              const float* __restrict__ bd,    // b[d]           [512]      f32
              float* __restrict__ dst)         // next state     [batch,512] f32
{
    __shared__ __bf16 ldsA[2][BM * LDH];   // 2 x 20480 B
    __shared__ __bf16 ldsB[2][BN * LDH];   // 2 x 10240 B   (total 60 KB)

    const int tid  = threadIdx.x;
    const int lane = tid & 31;
    const int wave = tid >> 5;     // 0..15
    const int wm   = wave & 3;     // 4 wave rows  (M, 64 each)
    const int wn   = wave >> 2;    // 4 wave cols  (N, 32 each)

    const int i0 = blockIdx.x * BM;
    const int j0 = blockIdx.y * BN;

    // ---- staging maps -------------------------------------------------------
    // A: 256 rows x 32 K = 8192 f32 / 512 thr = 16 f32 each (4x float4)
    const int aRow  = tid >> 1;            // 0..255
    const int aHalf = (tid & 1) << 4;      // 0 or 16
    // B: 128 rows x 32 K = 4096 f32 / 512 thr = 8 f32 each (2x float4)
    const int bRow  = tid >> 2;            // 0..127
    const int bQrt  = (tid & 3) << 3;      // 0,8,16,24

    const float* gA = src + (size_t)(i0 + aRow) * CASIZE + aHalf;
    const float* gB = Wd  + (size_t)(j0 + bRow) * CASIZE + bQrt;
    const int ldsOffA = aRow * LDH + aHalf;
    const int ldsOffB = bRow * LDH + bQrt;

    float4 ra0, ra1, ra2, ra3, rb0, rb1;

    auto stage = [&](int kt) {
        const float* pa = gA + kt * BK;
        const float* pb = gB + kt * BK;
        ra0 = *(const float4*)(pa + 0);  ra1 = *(const float4*)(pa + 4);
        ra2 = *(const float4*)(pa + 8);  ra3 = *(const float4*)(pa + 12);
        rb0 = *(const float4*)(pb + 0);  rb1 = *(const float4*)(pb + 4);
    };
    auto commit = [&](int buf) {           // f32 -> bf16 once, then LDS
        *(v8bf*)&ldsA[buf][ldsOffA + 0] = cvt8(ra0, ra1);
        *(v8bf*)&ldsA[buf][ldsOffA + 8] = cvt8(ra2, ra3);
        *(v8bf*)&ldsB[buf][ldsOffB]     = cvt8(rb0, rb1);
    };

    // ---- accumulators: wave owns 64x32 = 4x2 tiles of 16x16 -----------------
    v8f acc[4][2];
#pragma unroll
    for (int mt = 0; mt < 4; ++mt)
#pragma unroll
        for (int nt = 0; nt < 2; ++nt) {
            v8f z = {0.f, 0.f, 0.f, 0.f, 0.f, 0.f, 0.f, 0.f};
            acc[mt][nt] = z;
        }

    const int fr = lane & 15;    // row (A) / col (B) within 16x16 tile
    const int kh = lane >> 4;    // K half selector

    stage(0); commit(0);
    __syncthreads();

    int cur = 0;
#pragma unroll 1
    for (int kt = 0; kt < KTILES; ++kt) {
        if (kt + 1 < KTILES) stage(kt + 1);     // hide global latency

        // A fragments: lane m=fr; lanes 0-15: K {0..7,16..23}, lanes 16-31: +8
        v16bf afrag[4];
#pragma unroll
        for (int mt = 0; mt < 4; ++mt) {
            const __bf16* p = &ldsA[cur][(wm * 64 + mt * 16 + fr) * LDH];
            v8bf lo = *(const v8bf*)(p + kh * 8);
            v8bf hi = *(const v8bf*)(p + 16 + kh * 8);
            afrag[mt] = __builtin_shufflevector(lo, hi,
                0,1,2,3,4,5,6,7,8,9,10,11,12,13,14,15);
        }
        // B fragments: lane n=fr; lanes 0-15: K {0..7,16..23}, lanes 16-31: +8
        v16bf bfrag[2];
#pragma unroll
        for (int nt = 0; nt < 2; ++nt) {
            const __bf16* p = &ldsB[cur][(wn * 32 + nt * 16 + fr) * LDH + kh * 8];
            v8bf lo = *(const v8bf*)(p);
            v8bf hi = *(const v8bf*)(p + 16);
            bfrag[nt] = __builtin_shufflevector(lo, hi,
                0,1,2,3,4,5,6,7,8,9,10,11,12,13,14,15);
        }

#pragma unroll
        for (int mt = 0; mt < 4; ++mt)
#pragma unroll
            for (int nt = 0; nt < 2; ++nt)
                acc[mt][nt] = __builtin_amdgcn_wmma_f32_16x16x32_bf16(
                    false, afrag[mt], false, bfrag[nt],
                    (short)0, acc[mt][nt], false, false);

        if (kt + 1 < KTILES) {
            commit(cur ^ 1);
            __syncthreads();
            cur ^= 1;
        }
    }

    // ---- fused epilogue: bias, sigmoid, rule-110 gate, store ----------------
    // C layout: VGPR p -> M = p (+8 for lanes 16-31), N = lane%16
    const int moff = (lane >> 4) << 3;
#pragma unroll
    for (int mt = 0; mt < 4; ++mt) {
        const int ibase = i0 + wm * 64 + mt * 16 + moff;
#pragma unroll
        for (int nt = 0; nt < 2; ++nt) {
            const int j   = j0 + wn * 32 + nt * 16 + fr;
            const int jm1 = (j + CASIZE - 1) & (CASIZE - 1);
            const int jp1 = (j + 1) & (CASIZE - 1);
            const float bias = bd[j];
#pragma unroll
            for (int p = 0; p < 8; ++p) {
                const size_t row = (size_t)(ibase + p) * CASIZE;
                const float z = acc[mt][nt][p] + bias;
                const float m = 1.0f / (1.0f + __expf(-z));
                const float c = src[row + j];
                const float l = src[row + jm1];
                const float r = src[row + jp1];
                const int   nb  = (int)(l * 4.0f + c * 2.0f + r);
                const float bit = (float)((RULE >> nb) & 1);
                dst[row + j] = m * bit + (1.0f - m) * c;
            }
        }
    }
}

extern "C" void kernel_launch(void* const* d_in, const int* in_sizes, int n_in,
                              void* d_out, int out_size, void* d_ws, size_t ws_size,
                              hipStream_t stream)
{
    const float* x = (const float*)d_in[0];
    const float* W = (const float*)d_in[1];   // [8,512,512]
    const float* b = (const float*)d_in[2];   // [8,512]
    float* out = (float*)d_out;
    float* tmp = (float*)d_ws;                // one ping-pong state buffer

    const int batch = in_sizes[0] / CASIZE;   // 65536
    dim3 grid(batch / BM, CASIZE / BN);       // (256, 4)
    dim3 block(512);                          // 16 wave32s

    const float* cur = x;
    for (int d = 0; d < DEPTH; ++d) {
        float* nxt = (d & 1) ? out : tmp;     // d=7 (last) -> d_out
        aca_step<<<grid, block, 0, stream>>>(
            cur,
            W + (size_t)d * CASIZE * CASIZE,
            b + (size_t)d * CASIZE,
            nxt);
        cur = nxt;
    }
}
